// GNN_69535520522427
// MI455X (gfx1250) — matrix operand order repaired and checked by reading, not compiled, not run on hardware
//
#include <hip/hip_runtime.h>

// ---------------------------------------------------------------------------
// GraphSAGE 3-layer GNN for MI455X (gfx1250, wave32, WMMA)
// ---------------------------------------------------------------------------

typedef _Float16 v16h __attribute__((ext_vector_type(16)));
typedef _Float16 v8h  __attribute__((ext_vector_type(8)));
typedef float    v8f  __attribute__((ext_vector_type(8)));

#define N_NODES 50000
#define N_EDGES 800000

// ---------------- degree: deg[dst] += 1 ------------------------------------
__global__ void deg_kernel(const int* __restrict__ dst, float* __restrict__ deg, int E) {
    int e = blockIdx.x * blockDim.x + threadIdx.x;
    if (e < E) atomicAdd(&deg[dst[e]], 1.0f);
}

// ---------------- aggregation: msg[dst] += h[src] (float4 per thread) ------
__global__ void agg_kernel(const float* __restrict__ h, const int* __restrict__ src,
                           const int* __restrict__ dst, float* __restrict__ msg,
                           int E, int C) {
    long long t = (long long)blockIdx.x * blockDim.x + threadIdx.x;
    int chunks = C >> 2;                       // C is a multiple of 4
    long long total = (long long)E * chunks;
    if (t >= total) return;
    int e = (int)(t / chunks);
    int c = ((int)(t % chunks)) << 2;
    float4 v = *(const float4*)(h + (long long)src[e] * C + c);
    float* p = msg + (long long)dst[e] * C + c;
    atomicAdd(p + 0, v.x);
    atomicAdd(p + 1, v.y);
    atomicAdd(p + 2, v.z);
    atomicAdd(p + 3, v.w);
}

// ---------------- mean + f16 conversion of both GEMM operands --------------
__global__ void prep_kernel(const float* __restrict__ h, const float* __restrict__ msg,
                            const float* __restrict__ deg,
                            _Float16* __restrict__ in16, _Float16* __restrict__ mean16,
                            int N, int C) {
    int i = blockIdx.x * blockDim.x + threadIdx.x;
    if (i >= N * C) return;
    int row = i / C;
    float d = deg[row];
    d = d > 1.0f ? d : 1.0f;
    mean16[i] = (_Float16)(msg[i] / d);
    in16[i]   = (_Float16)h[i];
}

// ---------------- weight convert + transpose: W[K][F] -> WT[Fpad][K] f16 ---
__global__ void wconv_kernel(const float* __restrict__ W, _Float16* __restrict__ WT,
                             int K, int F, int Fpad) {
    int i = blockIdx.x * blockDim.x + threadIdx.x;
    if (i >= Fpad * K) return;
    int f = i / K;
    int k = i - f * K;
    WT[(long long)f * K + k] = (f < F) ? (_Float16)W[(long long)k * F + f] : (_Float16)0.0f;
}

// ---------------- fused SAGE GEMM: out = mean@Wl + x@Wr + b ----------------
// One wave per 16x16 output tile. K in steps of 32, two WMMAs per step.
// mode: 0 = linear, 1 = relu, 2 = relu + dropout(p=0.5, x2 scale, hash RNG)
__global__ void __launch_bounds__(32)
sage_gemm_wmma(const _Float16* __restrict__ in16,
               const _Float16* __restrict__ mean16,
               const _Float16* __restrict__ WTl,
               const _Float16* __restrict__ WTr,
               const float* __restrict__ bias,
               float* __restrict__ out,
               int K, int F, int ldout, int mode, unsigned seed) {
    const int lane = threadIdx.x;               // 0..31
    const int m0   = blockIdx.x * 16;
    const int f0   = blockIdx.y * 16;
    const int r    = lane & 15;
    const bool lo  = lane < 16;
    const int kh   = lo ? 0 : 16;               // A-fragment K sub-offset
    const int kb   = lo ? 0 : 16;               // B-fragment K sub-offset
    const int row  = m0 + r;
    const int col  = f0 + r;

    // accumulator init = bias broadcast (all 8 VGPRs share column `col`)
    float bv = (col < F) ? bias[col] : 0.0f;
    v8f acc;
#pragma unroll
    for (int i = 0; i < 8; ++i) acc[i] = bv;

    const _Float16* pAm = mean16 + (long long)row * K + kh;
    const _Float16* pAx = in16   + (long long)row * K + kh;
    const _Float16* pBl = WTl    + (long long)col * K + kb;  // WT padded to Fpad rows
    const _Float16* pBr = WTr    + (long long)col * K + kb;

    for (int kk = 0; kk < K; kk += 32) {
        // A fragments: halves {kk+kh .. +7} and {kk+8+kh .. +7}
        v8h am0 = *(const v8h*)(pAm + kk);
        v8h am1 = *(const v8h*)(pAm + kk + 8);
        v8h ax0 = *(const v8h*)(pAx + kk);
        v8h ax1 = *(const v8h*)(pAx + kk + 8);
        v16h am, ax;
#pragma unroll
        for (int i = 0; i < 8; ++i) { am[i] = am0[i]; am[i + 8] = am1[i]; }
#pragma unroll
        for (int i = 0; i < 8; ++i) { ax[i] = ax0[i]; ax[i + 8] = ax1[i]; }
        // B fragments: 16 contiguous K values per lane from transposed weights
        v16h bl = *(const v16h*)(pBl + kk);
        v16h br = *(const v16h*)(pBr + kk);

        acc = __builtin_amdgcn_wmma_f32_16x16x32_f16(false, am, false, bl,
                                                     (short)0, acc, false, false);
        acc = __builtin_amdgcn_wmma_f32_16x16x32_f16(false, ax, false, br,
                                                     (short)0, acc, false, false);
    }

    if (col < F) {
#pragma unroll
        for (int i = 0; i < 8; ++i) {
            int m = m0 + i + (lo ? 0 : 8);
            float v = acc[i];
            if (mode >= 1) v = v > 0.0f ? v : 0.0f;
            if (mode == 2) {
                unsigned hsh = ((unsigned)(m * F + col)) ^ seed;
                hsh *= 2654435761u; hsh ^= hsh >> 16;
                hsh *= 2246822519u; hsh ^= hsh >> 13;
                v = (hsh & 1u) ? v * 2.0f : 0.0f;
            }
            out[(long long)m * ldout + col] = v;
        }
    }
}

// ---------------- log_softmax over F=300 per node --------------------------
__global__ void logsoftmax_kernel(const float* __restrict__ logits,
                                  float* __restrict__ out, int N, int F) {
    int n = blockIdx.x * blockDim.x + threadIdx.x;
    if (n >= N) return;
    const float* p = logits + (long long)n * F;
    float mx = p[0];
    for (int j = 1; j < F; ++j) mx = fmaxf(mx, p[j]);
    float s = 0.0f;
    for (int j = 0; j < F; ++j) s += __expf(p[j] - mx);
    float ls = __logf(s);
    float* o = out + (long long)n * F;
    for (int j = 0; j < F; ++j) o[j] = p[j] - mx - ls;
}

// ---------------------------------------------------------------------------
extern "C" void kernel_launch(void* const* d_in, const int* in_sizes, int n_in,
                              void* d_out, int out_size, void* d_ws, size_t ws_size,
                              hipStream_t stream) {
    const float* x    = (const float*)d_in[0];     // [N,128]
    const int*   edge = (const int*)d_in[1];       // [2,E]
    const int E = in_sizes[1] / 2;
    const int* src = edge;
    const int* dst = edge + E;
    const float* Wl1 = (const float*)d_in[2];
    const float* Wr1 = (const float*)d_in[3];
    const float* b1  = (const float*)d_in[4];
    const float* Wl2 = (const float*)d_in[5];
    const float* Wr2 = (const float*)d_in[6];
    const float* b2  = (const float*)d_in[7];
    const float* Wl3 = (const float*)d_in[8];
    const float* Wr3 = (const float*)d_in[9];
    const float* b3  = (const float*)d_in[10];
    float* out = (float*)d_out;

    // ---- workspace carve-out (256B aligned) ----
    char* ws = (char*)d_ws;
    size_t off = 0;
    auto carve = [&](size_t bytes) -> char* {
        char* p = ws + off;
        off += (bytes + 255) & ~(size_t)255;
        return p;
    };
    float*    deg    = (float*)   carve((size_t)N_NODES * 4);
    float*    msg    = (float*)   carve((size_t)N_NODES * 256 * 4);
    _Float16* in16   = (_Float16*)carve((size_t)N_NODES * 256 * 2);
    _Float16* mean16 = (_Float16*)carve((size_t)N_NODES * 256 * 2);
    float*    h1     = (float*)   carve((size_t)N_NODES * 256 * 4);
    float*    h2     = (float*)   carve((size_t)N_NODES * 256 * 4);
    float*    logits = (float*)   carve((size_t)N_NODES * 300 * 4);
    _Float16* WTl    = (_Float16*)carve((size_t)304 * 256 * 2);
    _Float16* WTr    = (_Float16*)carve((size_t)304 * 256 * 2);

    const int TPB = 256;
    auto blocks = [](long long n, int tpb) { return (unsigned)((n + tpb - 1) / tpb); };

    // degrees (shared across layers)
    hipMemsetAsync(deg, 0, (size_t)N_NODES * 4, stream);
    deg_kernel<<<blocks(E, TPB), TPB, 0, stream>>>(dst, deg, E);

    // ======================= Layer 1: 128 -> 256 (relu+dropout) ============
    {
        const int C = 128, F = 256;
        hipMemsetAsync(msg, 0, (size_t)N_NODES * C * 4, stream);
        agg_kernel<<<blocks((long long)E * (C >> 2), TPB), TPB, 0, stream>>>(
            x, src, dst, msg, E, C);
        prep_kernel<<<blocks((long long)N_NODES * C, TPB), TPB, 0, stream>>>(
            x, msg, deg, in16, mean16, N_NODES, C);
        wconv_kernel<<<blocks((long long)F * C, TPB), TPB, 0, stream>>>(Wl1, WTl, C, F, F);
        wconv_kernel<<<blocks((long long)F * C, TPB), TPB, 0, stream>>>(Wr1, WTr, C, F, F);
        dim3 grid(N_NODES / 16, F / 16);
        sage_gemm_wmma<<<grid, 32, 0, stream>>>(in16, mean16, WTl, WTr, b1, h1,
                                                C, F, F, /*mode=*/2, 0x9E3779B1u);
    }

    // ======================= Layer 2: 256 -> 256 (relu+dropout) ============
    {
        const int C = 256, F = 256;
        hipMemsetAsync(msg, 0, (size_t)N_NODES * C * 4, stream);
        agg_kernel<<<blocks((long long)E * (C >> 2), TPB), TPB, 0, stream>>>(
            h1, src, dst, msg, E, C);
        prep_kernel<<<blocks((long long)N_NODES * C, TPB), TPB, 0, stream>>>(
            h1, msg, deg, in16, mean16, N_NODES, C);
        wconv_kernel<<<blocks((long long)F * C, TPB), TPB, 0, stream>>>(Wl2, WTl, C, F, F);
        wconv_kernel<<<blocks((long long)F * C, TPB), TPB, 0, stream>>>(Wr2, WTr, C, F, F);
        dim3 grid(N_NODES / 16, F / 16);
        sage_gemm_wmma<<<grid, 32, 0, stream>>>(in16, mean16, WTl, WTr, b2, h2,
                                                C, F, F, /*mode=*/2, 0x85EBCA77u);
    }

    // ======================= Layer 3: 256 -> 300 (linear) ==================
    {
        const int C = 256, F = 300, Fpad = 304;
        hipMemsetAsync(msg, 0, (size_t)N_NODES * C * 4, stream);
        agg_kernel<<<blocks((long long)E * (C >> 2), TPB), TPB, 0, stream>>>(
            h2, src, dst, msg, E, C);
        prep_kernel<<<blocks((long long)N_NODES * C, TPB), TPB, 0, stream>>>(
            h2, msg, deg, in16, mean16, N_NODES, C);
        wconv_kernel<<<blocks((long long)Fpad * C, TPB), TPB, 0, stream>>>(Wl3, WTl, C, F, Fpad);
        wconv_kernel<<<blocks((long long)Fpad * C, TPB), TPB, 0, stream>>>(Wr3, WTr, C, F, Fpad);
        dim3 grid(N_NODES / 16, Fpad / 16);   // 19 f-tiles, stores guarded at F=300
        sage_gemm_wmma<<<grid, 32, 0, stream>>>(in16, mean16, WTl, WTr, b3, logits,
                                                C, F, F, /*mode=*/0, 0u);
    }

    logsoftmax_kernel<<<blocks(N_NODES, 128), 128, 0, stream>>>(logits, out, N_NODES, 300);
}
